// tb_module_1x1_4939212390875
// MI455X (gfx1250) — 8-bit matrix operand order repaired and checked by reading, compiled, not run on hardware
//
#include <hip/hip_runtime.h>

typedef __attribute__((ext_vector_type(8))) int v8i;

#define IC 256
#define OC 256
#define ROWS_PER_BLOCK 128

__global__ __launch_bounds__(256) void pack_weights_kernel(
    const float* __restrict__ w, unsigned* __restrict__ bpack) {
  int idx  = blockIdx.x * 256 + threadIdx.x;
  int v    = idx & 7;
  int lane = (idx >> 3) & 31;
  int t    = (idx >> 8) & 15;
  int s    = (idx >> 12) & 3;
  int col  = t * 16 + (lane & 15);
  int half = lane >> 4;
  unsigned d = 0;
#pragma unroll
  for (int b = 0; b < 4; ++b) {
    int hwK = (v >> 2) * 32 + half * 16 + (v & 3) * 4 + b;
    int a = hwK >> 4;
    int r = hwK & 15;
    int m = half * 32 + v * 4 + b;
    int k = s * 64 + m;
    int wi = (int)w[k * OC + col];
    d |= ((unsigned)(wi & 0xff)) << (8 * b);
  }
  bpack[idx] = d;
}

__device__ __forceinline__ void requant_store(
    const v8i& acc, int t, int laneMod, int laneDiv, int rowbase,
    const float* __restrict__ bias, const float* __restrict__ scale,
    float* __restrict__ out) {
  const int cb = t * 16 + laneMod;
  const int bi = (int)bias[cb];
  const int si = (int)scale[cb];
  float* op = out + (size_t)(rowbase + laneDiv * 8) * OC + cb;
#pragma unroll
  for (int r = 0; r < 8; ++r) {
    int vv = ((acc[r] + bi) * si + 128) >> 8;
    vv = vv < 0 ? 0 : (vv > 127 ? 127 : vv);
    op[(size_t)r * OC] = (float)vv;
  }
}

__global__ __launch_bounds__(256) void conv1x1_wmma_kernel(
    const float* __restrict__ feats, const unsigned* __restrict__ bpack,
    const float* __restrict__ bias, const float* __restrict__ scale,
    float* __restrict__ out, int nrows) {
  __shared__ unsigned lds_b[16384];

  const int tid = threadIdx.x;
  {
    const int4* gb = (const int4*)bpack;
    int4* lb = (int4*)lds_b;
#pragma unroll
    for (int j = 0; j < 16; ++j) lb[tid + j * 256] = gb[tid + j * 256];
  }
  __syncthreads();

  const int wave    = tid >> 5;
  const int lane    = tid & 31;
  const int laneMod = lane & 15;
  const int laneDiv = lane >> 4;
  const int rowbase = blockIdx.x * ROWS_PER_BLOCK + wave * 16;
  if (rowbase + 16 > nrows) return;

  v8i A[4];
  const float* arow = feats + (size_t)(rowbase + laneMod) * IC + laneDiv * 32;
#pragma unroll
  for (int s = 0; s < 4; ++s) {
    const float4* p = (const float4*)(arow + s * 64);
#pragma unroll
    for (int q = 0; q < 8; ++q) {
      float4 f = p[q];
      int b0 = (int)f.x, b1 = (int)f.y, b2 = (int)f.z, b3 = (int)f.w;
      A[s][q] = (b0 & 0xff) | ((b1 & 0xff) << 8) | ((b2 & 0xff) << 16) |
                ((b3 & 0xff) << 24);
    }
  }

  for (int t = 0; t < 16; t += 2) {
    v8i acc0 = {0, 0, 0, 0, 0, 0, 0, 0};
    v8i acc1 = {0, 0, 0, 0, 0, 0, 0, 0};
#pragma unroll
    for (int s = 0; s < 4; ++s) {
      const int4* bp0 = (const int4*)(&lds_b[(((s * 16 + t) * 32) + lane) * 8]);
      const int4* bp1 =
          (const int4*)(&lds_b[(((s * 16 + t + 1) * 32) + lane) * 8]);
      int4 q0 = bp0[0], q1 = bp0[1];
      int4 q2 = bp1[0], q3 = bp1[1];
      v8i B0, B1;
      B0[0] = q0.x; B0[1] = q0.y; B0[2] = q0.z; B0[3] = q0.w;
      B0[4] = q1.x; B0[5] = q1.y; B0[6] = q1.z; B0[7] = q1.w;
      B1[0] = q2.x; B1[1] = q2.y; B1[2] = q2.z; B1[3] = q2.w;
      B1[4] = q3.x; B1[5] = q3.y; B1[6] = q3.z; B1[7] = q3.w;
      acc0 = __builtin_amdgcn_wmma_i32_16x16x64_iu8(true, A[s], true, B0, acc0,
                                                    false, false);
      acc1 = __builtin_amdgcn_wmma_i32_16x16x64_iu8(true, A[s], true, B1, acc1,
                                                    false, false);
    }
    requant_store(acc0, t, laneMod, laneDiv, rowbase, bias, scale, out);
    requant_store(acc1, t + 1, laneMod, laneDiv, rowbase, bias, scale, out);
  }
}

extern "C" void kernel_launch(void* const* d_in, const int* in_sizes, int n_in,
                              void* d_out, int out_size, void* d_ws,
                              size_t ws_size, hipStream_t stream) {
  const float* feats  = (const float*)d_in[0];
  const float* weight = (const float*)d_in[1];
  const float* bias   = (const float*)d_in[2];
  const float* scale  = (const float*)d_in[3];
  float* out = (float*)d_out;
  unsigned* bpack = (unsigned*)d_ws;
  int nrows = in_sizes[0] / IC;

  pack_weights_kernel<<<64, 256, 0, stream>>>(weight, bpack);
  int blocks = (nrows + ROWS_PER_BLOCK - 1) / ROWS_PER_BLOCK;
  conv1x1_wmma_kernel<<<blocks, 256, 0, stream>>>(feats, bpack, bias, scale,
                                                  out, nrows);
}
